// QuantizedMoEExpert_27874337751307
// MI455X (gfx1250) — matrix operand order repaired and checked by reading, not compiled, not run on hardware
//
#include <hip/hip_runtime.h>

// ---- vector types ----
typedef __attribute__((ext_vector_type(16))) _Float16 v16h;
typedef __attribute__((ext_vector_type(8)))  _Float16 v8h;
typedef __attribute__((ext_vector_type(2)))  _Float16 v2h;
typedef __attribute__((ext_vector_type(8)))  float    v8f;
typedef __attribute__((ext_vector_type(4)))  float    f32x4;
typedef __attribute__((ext_vector_type(4)))  int      i32x4;

#define TOKENS 8192
#define IN_F   4096
#define OUT_F  14336
#define BM 128
#define BN 128
#define BK 32
#define NKITER (IN_F / BK)          // 128
#define LDW 5                       // LDS row stride in v8h units (40 halves) -> bank-conflict-free

// Decode 4 int32s (one byte each = 8 nibbles) into 8 f16 values (q - z).
// Magic bias: (0x6400 | q) as f16 == 1024 + q exactly; subtracting 1024.0h is
// exact, so only z keeps its (inherent) f16 rounding. Two v_pk_sub_f16 per int.
__device__ __forceinline__ v8h decode8(i32x4 w, v2h zz) {
    const v2h k1024 = {(_Float16)1024.0f, (_Float16)1024.0f};
    i32x4 hv;
#pragma unroll
    for (int j = 0; j < 4; ++j) {
        unsigned u = (unsigned)w[j];
        unsigned t = ((u & 0xFu) | ((u & 0xF0u) << 12)) | 0x64006400u;
        v2h h = __builtin_bit_cast(v2h, t);
        h = h - k1024;   // exact: yields (q_lo, q_hi)
        h = h - zz;      // packed zero-point subtract
        hv[j] = __builtin_bit_cast(int, h);
    }
    return __builtin_bit_cast(v8h, hv);
}

// Convert 8 consecutive f32 (two f32x4) to v8h -> v_cvt_pk_f16_f32 pairs.
__device__ __forceinline__ v8h cvt8(f32x4 a, f32x4 b) {
    v8h h;
#pragma unroll
    for (int j = 0; j < 4; ++j) h[j]     = (_Float16)a[j];
#pragma unroll
    for (int j = 0; j < 4; ++j) h[4 + j] = (_Float16)b[j];
    return h;
}

__global__ __launch_bounds__(256)
void q4_gemm_wmma_kernel(const float* __restrict__ x,
                         const int*   __restrict__ pw,      // [OUT_F, IN_F/2]
                         const float* __restrict__ scales,  // [OUT_F]
                         const float* __restrict__ zps,     // [OUT_F]
                         float*       __restrict__ out)     // [TOKENS, OUT_F]
{
    // Double-buffered tiles: one barrier per K-iteration.
    __shared__ v8h ldsA[2][BM * LDW];   // 2 x 10 KB : x tile as f16  [m][k]
    __shared__ v8h ldsB[2][BN * LDW];   // 2 x 10 KB : (q-z) as f16   [n][k]

    const int tid  = threadIdx.x;       // 0..255
    const int lane = tid & 31;
    const int wave = tid >> 5;          // 0..7
    const int wm   = wave >> 2;         // 0..1 : 64-row M slab
    const int wn   = wave & 3;          // 0..3 : 32-col N slab
    const int lh   = lane & 15;
    const int lsel = lane >> 4;

    // M-tiles vary fastest: consecutive blocks share one W n-tile and stream x,
    // which (128 MB) stays resident in the 192 MB L2.
    const int m0 = blockIdx.x * BM;
    const int n0 = blockIdx.y * BN;

    // ---- staging assignment: 2 threads per tile row ----
    const int row = tid >> 1;           // 0..127
    const int seg = tid & 1;

    const float* aSrc = x  + (size_t)(m0 + row) * IN_F     + seg * 16; // 16 floats
    const int*   bSrc = pw + (size_t)(n0 + row) * (IN_F/2) + seg * 8;  // 8 ints
    const _Float16 zph = (_Float16)zps[n0 + row];
    const v2h zz = {zph, zph};

    f32x4 aReg[4];
    i32x4 bReg[2];

    v8f acc[4][2];
#pragma unroll
    for (int mi = 0; mi < 4; ++mi)
#pragma unroll
        for (int ni = 0; ni < 2; ++ni)
            acc[mi][ni] = (v8f){0.f, 0.f, 0.f, 0.f, 0.f, 0.f, 0.f, 0.f};

    // ---- prologue: tile 0 -> buf0, stage tile 1 into regs ----
#pragma unroll
    for (int i = 0; i < 4; ++i) aReg[i] = ((const f32x4*)aSrc)[i];
#pragma unroll
    for (int i = 0; i < 2; ++i) bReg[i] = ((const i32x4*)bSrc)[i];

    ldsA[0][row * LDW + seg * 2 + 0] = cvt8(aReg[0], aReg[1]);
    ldsA[0][row * LDW + seg * 2 + 1] = cvt8(aReg[2], aReg[3]);
    ldsB[0][row * LDW + seg * 2 + 0] = decode8(bReg[0], zz);
    ldsB[0][row * LDW + seg * 2 + 1] = decode8(bReg[1], zz);

    aSrc += BK;  bSrc += BK / 2;
#pragma unroll
    for (int i = 0; i < 4; ++i) aReg[i] = ((const f32x4*)aSrc)[i];
#pragma unroll
    for (int i = 0; i < 2; ++i) bReg[i] = ((const i32x4*)bSrc)[i];

    __syncthreads();

    for (int kk = 0; kk < NKITER; ++kk) {
        const int cur = kk & 1;
        const int nxt = cur ^ 1;

        // ---- write staged regs (tile kk+1) into the other buffer ----
        if (kk + 1 < NKITER) {
            ldsA[nxt][row * LDW + seg * 2 + 0] = cvt8(aReg[0], aReg[1]);
            ldsA[nxt][row * LDW + seg * 2 + 1] = cvt8(aReg[2], aReg[3]);
            ldsB[nxt][row * LDW + seg * 2 + 0] = decode8(bReg[0], zz);
            ldsB[nxt][row * LDW + seg * 2 + 1] = decode8(bReg[1], zz);
        }
        // ---- issue global loads for tile kk+2 (overlaps the WMMA burst) ----
        if (kk + 2 < NKITER) {
            aSrc += BK;  bSrc += BK / 2;
#pragma unroll
            for (int i = 0; i < 4; ++i) aReg[i] = ((const f32x4*)aSrc)[i];
#pragma unroll
            for (int i = 0; i < 2; ++i) bReg[i] = ((const i32x4*)bSrc)[i];
            __builtin_prefetch(aSrc + 4 * BK, 0, 1);         // global_prefetch_b8
            __builtin_prefetch(bSrc + 4 * (BK / 2), 0, 1);
        }

        // ---- load ALL fragments first (ds_load_b128 burst) ----
        v16h bfrag[2], afrag[4];
#pragma unroll
        for (int ni = 0; ni < 2; ++ni) {
            const int br = wn * 32 + ni * 16 + lh;
            v8h b0 = ldsB[cur][br * LDW + lsel + 0];       // K[8*lsel .. +8)
            v8h b1 = ldsB[cur][br * LDW + lsel + 2];       // K[8*lsel+16 .. +8)
            bfrag[ni] = __builtin_shufflevector(b0, b1,
                0, 1, 2, 3, 4, 5, 6, 7, 8, 9, 10, 11, 12, 13, 14, 15);
        }
#pragma unroll
        for (int mi = 0; mi < 4; ++mi) {
            const int ar = wm * 64 + mi * 16 + lh;
            v8h a0 = ldsA[cur][ar * LDW + lsel + 0];       // K[8*lsel .. +8)
            v8h a1 = ldsA[cur][ar * LDW + lsel + 2];       // K[8*lsel+16 .. +8)
            afrag[mi] = __builtin_shufflevector(a0, a1,
                0, 1, 2, 3, 4, 5, 6, 7, 8, 9, 10, 11, 12, 13, 14, 15);
        }

        // ---- 8 back-to-back WMMAs ----
#pragma unroll
        for (int mi = 0; mi < 4; ++mi) {
            acc[mi][0] = __builtin_amdgcn_wmma_f32_16x16x32_f16(
                false, afrag[mi], false, bfrag[0], (short)0, acc[mi][0], false, false);
            acc[mi][1] = __builtin_amdgcn_wmma_f32_16x16x32_f16(
                false, afrag[mi], false, bfrag[1], (short)0, acc[mi][1], false, false);
        }

        __syncthreads();   // single barrier per iteration (double buffer)
    }

    // ---- epilogue: apply per-channel scale, store f32 ----
#pragma unroll
    for (int ni = 0; ni < 2; ++ni) {
        const int nc = n0 + wn * 32 + ni * 16 + lh;
        const float s = scales[nc];
#pragma unroll
        for (int mi = 0; mi < 4; ++mi) {
            const int mr = m0 + wm * 64 + mi * 16 + 8 * lsel;
            float* o = out + (size_t)mr * OUT_F + nc;
#pragma unroll
            for (int r = 0; r < 8; ++r)
                o[(size_t)r * OUT_F] = acc[mi][ni][r] * s;
        }
    }
}

extern "C" void kernel_launch(void* const* d_in, const int* in_sizes, int n_in,
                              void* d_out, int out_size, void* d_ws, size_t ws_size,
                              hipStream_t stream) {
    const float* x      = (const float*)d_in[0];
    const int*   pw     = (const int*)d_in[1];
    const float* scales = (const float*)d_in[2];
    const float* zps    = (const float*)d_in[3];
    float*       out    = (float*)d_out;

    dim3 grid(TOKENS / BM, OUT_F / BN);   // (64, 112); M fastest for L2 reuse of x
    q4_gemm_wmma_kernel<<<grid, 256, 0, stream>>>(x, pw, scales, zps, out);
}
